// HungarianLoss_14078902797065
// MI455X (gfx1250) — matrix operand order repaired and checked by reading, not compiled, not run on hardware
//
#include <hip/hip_runtime.h>
#include <hip/hip_bf16.h>

// Problem constants (match reference)
#define B_   32
#define NS_  1000
#define NT_  100
#define C_   256
#define SP_  1024   // padded source stride in costT
#define TP_  112    // padded target dim (7 tiles of 16)

typedef __attribute__((ext_vector_type(16))) _Float16 v16h;
typedef __attribute__((ext_vector_type(8)))  float    v8f;
typedef __attribute__((ext_vector_type(4)))  float    v4f;

// ---------------------------------------------------------------------------
// Kernel 0: logsumexp over C=256 per (b,s) row.  One wave32 per row.
// ---------------------------------------------------------------------------
__global__ void hl_lse_kernel(const float* __restrict__ logits,
                              float* __restrict__ lse) {
  int wid  = (int)((blockIdx.x * blockDim.x + threadIdx.x) >> 5);
  int lane = (int)(threadIdx.x & 31);
  if (wid >= B_ * NS_) return;
  const float* row = logits + (size_t)wid * C_;
  float vals[8];
  float mx = -1e30f;
#pragma unroll
  for (int k = 0; k < 8; ++k) {
    vals[k] = row[lane + 32 * k];
    mx = fmaxf(mx, vals[k]);
  }
#pragma unroll
  for (int off = 16; off; off >>= 1) mx = fmaxf(mx, __shfl_xor(mx, off, 32));
  float sum = 0.0f;
#pragma unroll
  for (int k = 0; k < 8; ++k) sum += __expf(vals[k] - mx);
#pragma unroll
  for (int off = 16; off; off >>= 1) sum += __shfl_xor(sum, off, 32);
  if (lane == 0) {
    int b = wid / NS_, s = wid % NS_;
    lse[b * SP_ + s] = mx + __logf(sum);
  }
}

// ---------------------------------------------------------------------------
// Kernel 1: pairwise cost via WMMA (picked-logit = logits x one-hot GEMM)
// grid = (63 s-tiles, 32 batches); block = 224 (7 wave32s, one t-tile each).
// Writes costT[b][t][s] (transposed for coalesced Hungarian row scans).
// ---------------------------------------------------------------------------
__global__ void hl_cost_kernel(const float* __restrict__ logits,
                               const int*   __restrict__ tcls,
                               const float* __restrict__ sboxg,
                               const float* __restrict__ tboxg,
                               const float* __restrict__ lse,
                               float* __restrict__ costT) {
  const int stile = (int)blockIdx.x;       // 0..62
  const int b     = (int)blockIdx.y;       // 0..31
  const int wave  = (int)(threadIdx.x >> 5);  // 0..6  -> t tile
  const int lane  = (int)(threadIdx.x & 31);
  const int s0 = stile * 16;
  const int t0 = wave * 16;
  const int r  = lane & 15;
  const int h  = lane >> 4;

  // B/C/D column for this lane
  const int t = t0 + r;
  const bool tvalid = (t < NT_);
  const int cls = tvalid ? tcls[b * NT_ + t] : -1;

  // A row for this lane (clamped; invalid rows never stored)
  int sA = s0 + r; if (sA > NS_ - 1) sA = NS_ - 1;
  const float* arow = logits + ((size_t)b * NS_ + sA) * C_;

  v8f acc = {};
#pragma unroll
  for (int kk = 0; kk < 8; ++kk) {
    const int kbase = kk * 32;
    // A layout (16-bit 16x32): e<8 -> K=kbase+8h+e ; e>=8 -> K=kbase+8h+16+(e-8)
    const float* p = arow + kbase + 8 * h;
    v4f f0 = *(const v4f*)(p);
    v4f f1 = *(const v4f*)(p + 4);
    v4f f2 = *(const v4f*)(p + 16);
    v4f f3 = *(const v4f*)(p + 20);
    v16h a;
    a[0]  = (_Float16)f0[0]; a[1]  = (_Float16)f0[1];
    a[2]  = (_Float16)f0[2]; a[3]  = (_Float16)f0[3];
    a[4]  = (_Float16)f1[0]; a[5]  = (_Float16)f1[1];
    a[6]  = (_Float16)f1[2]; a[7]  = (_Float16)f1[3];
    a[8]  = (_Float16)f2[0]; a[9]  = (_Float16)f2[1];
    a[10] = (_Float16)f2[2]; a[11] = (_Float16)f2[3];
    a[12] = (_Float16)f3[0]; a[13] = (_Float16)f3[1];
    a[14] = (_Float16)f3[2]; a[15] = (_Float16)f3[3];
    // One-hot B (32x16): lane = column; e<8 -> K=kbase+8h+e ; e>=8 -> K=kbase+8h+16+(e-8)
    const int rel = cls - kbase;
    v16h bb;
#pragma unroll
    for (int e = 0; e < 16; ++e)
      bb[e] = (rel == e + 8 * h + (e >= 8 ? 8 : 0)) ? (_Float16)1.0f : (_Float16)0.0f;
    acc = __builtin_amdgcn_wmma_f32_16x16x32_f16(
        false, a, false, bb, (short)0, acc, false, false);
  }
  if (!tvalid) return;

  // Epilogue: C/D lane L holds N=L&15 (t), M=v+8h (s).  Fuse CE+GIoU+L1.
  v4f tb = *(const v4f*)(tboxg + ((size_t)b * NT_ + t) * 4);
  const float tx1 = tb[0], ty1 = tb[1], tx2 = tb[2], ty2 = tb[3];
  const float area_t = (tx2 - tx1) * (ty2 - ty1);
  float* crow = costT + ((size_t)b * TP_ + t) * SP_;
#pragma unroll
  for (int v = 0; v < 8; ++v) {
    const int s = s0 + v + 8 * h;
    if (s >= NS_) continue;
    const float ce = lse[b * SP_ + s] - acc[v];
    v4f sb = *(const v4f*)(sboxg + ((size_t)b * NS_ + s) * 4);
    const float sx1 = sb[0], sy1 = sb[1], sx2 = sb[2], sy2 = sb[3];
    const float ix1 = fmaxf(sx1, tx1), iy1 = fmaxf(sy1, ty1);
    const float ix2 = fminf(sx2, tx2), iy2 = fminf(sy2, ty2);
    const float inter  = fmaxf(ix2 - ix1, 0.0f) * fmaxf(iy2 - iy1, 0.0f);
    const float area_s = (sx2 - sx1) * (sy2 - sy1);
    const float uni    = area_s + area_t - inter;
    const float iou    = inter / uni;
    const float cx1 = fminf(sx1, tx1), cy1 = fminf(sy1, ty1);
    const float cx2 = fmaxf(sx2, tx2), cy2 = fmaxf(sy2, ty2);
    const float car  = (cx2 - cx1) * (cy2 - cy1);
    const float giou = iou - (car - uni) / car;
    const float l1 = 0.25f * (fabsf(sx1 - tx1) + fabsf(sy1 - ty1) +
                              fabsf(sx2 - tx2) + fabsf(sy2 - ty2));
    crow[s] = ce + 10.0f * (1.0f - giou) + l1;
  }
}

// ---------------------------------------------------------------------------
// Kernel 2: Jonker-Volgenant shortest augmenting path (transposed: n=100
// target rows, m=1000 source columns).  One block of 1024 threads per batch;
// all dual/assignment state in LDS.  Epilogue: mean of matched costs.
// ---------------------------------------------------------------------------
__global__ __launch_bounds__(1024)
void hl_hungarian_kernel(const float* __restrict__ costT,
                         float* __restrict__ out) {
  const int b = (int)blockIdx.x;
  const int j = (int)threadIdx.x;          // column index; valid 1..NS_
  const int lane = j & 31, wid = j >> 5;
  const bool valid = (j >= 1 && j <= NS_);

  __shared__ float v_f[1024];
  __shared__ float minv[1024];
  __shared__ int   way_[1024];
  __shared__ int   p_[1024];
  __shared__ int   used[1024];
  __shared__ float u_[136];
  __shared__ float pv[32];
  __shared__ int   pi[32];
  __shared__ int   s_j0;
  __shared__ float s_delta;

  const float* cb = costT + (size_t)b * TP_ * SP_;

  v_f[j] = 0.0f; minv[j] = 0.0f; way_[j] = 0; p_[j] = 0; used[j] = 0;
  if (j < 136) u_[j] = 0.0f;
  __syncthreads();

  for (int i = 1; i <= NT_; ++i) {
    if (j == 0) { p_[0] = i; s_j0 = 0; }
    minv[j] = 1e30f;
    used[j] = 0;
    __syncthreads();
    int j0 = 0;
    while (true) {
      if (j == 0) used[j0] = 1;
      __syncthreads();
      const int   i0 = p_[j0];
      const float uu = u_[i0];
      const float* crow = cb + (size_t)(i0 - 1) * SP_;
      if (valid && !used[j]) {
        const float cur = crow[j - 1] - uu - v_f[j];
        if (cur < minv[j]) { minv[j] = cur; way_[j] = j0; }
      }
      __syncthreads();
      // argmin over free columns (tie-break: lowest index, matches np.argmin)
      float val = (valid && !used[j]) ? minv[j] : 1e30f;
      int   idx = j;
#pragma unroll
      for (int off = 16; off; off >>= 1) {
        const float ov = __shfl_xor(val, off, 32);
        const int   oi = __shfl_xor(idx, off, 32);
        if (ov < val || (ov == val && oi < idx)) { val = ov; idx = oi; }
      }
      if (lane == 0) { pv[wid] = val; pi[wid] = idx; }
      __syncthreads();
      if (wid == 0) {
        float v2 = pv[lane]; int i2 = pi[lane];
#pragma unroll
        for (int off = 16; off; off >>= 1) {
          const float ov = __shfl_xor(v2, off, 32);
          const int   oi = __shfl_xor(i2, off, 32);
          if (ov < v2 || (ov == v2 && oi < i2)) { v2 = ov; i2 = oi; }
        }
        if (lane == 0) { s_j0 = i2; s_delta = v2; }
      }
      __syncthreads();
      const int   j1    = s_j0;
      const float delta = s_delta;
      if (j <= NS_) {
        if (used[j]) { u_[p_[j]] += delta; v_f[j] -= delta; }  // distinct rows: no collision
        else if (valid) { minv[j] -= delta; }
      }
      __syncthreads();
      j0 = j1;
      if (p_[j0] == 0) break;
    }
    // augment along alternating path (serial, thread 0)
    if (j == 0) {
      int jj = s_j0;
      while (jj) { const int jp = way_[jj]; p_[jj] = p_[jp]; jj = jp; }
    }
    __syncthreads();
  }

  // mean of matched costs -> out[b]
  float val = 0.0f;
  if (valid && p_[j] > 0) val = cb[(size_t)(p_[j] - 1) * SP_ + (j - 1)];
#pragma unroll
  for (int off = 16; off; off >>= 1) val += __shfl_xor(val, off, 32);
  if (lane == 0) pv[wid] = val;
  __syncthreads();
  if (j == 0) {
    float s = 0.0f;
    for (int w = 0; w < 32; ++w) s += pv[w];
    out[b] = s / (float)NT_;
  }
}

// ---------------------------------------------------------------------------
extern "C" void kernel_launch(void* const* d_in, const int* in_sizes, int n_in,
                              void* d_out, int out_size, void* d_ws, size_t ws_size,
                              hipStream_t stream) {
  (void)in_sizes; (void)n_in; (void)out_size; (void)ws_size;
  const float* logits = (const float*)d_in[0];   // (B, NS, C) f32
  const int*   tcls   = (const int*)d_in[1];     // (B, NT) i32
  const float* sboxes = (const float*)d_in[2];   // (B, NS, 4) f32
  const float* tboxes = (const float*)d_in[3];   // (B, NT, 4) f32
  float* out = (float*)d_out;                    // (B,) f32

  float* lse   = (float*)d_ws;                   // B*SP_ floats
  float* costT = lse + (size_t)B_ * SP_;         // B*TP_*SP_ floats (~14.7MB)

  hl_lse_kernel<<<dim3((B_ * NS_ + 7) / 8), 256, 0, stream>>>(logits, lse);
  hl_cost_kernel<<<dim3(63, B_), 224, 0, stream>>>(logits, tcls, sboxes,
                                                   tboxes, lse, costT);
  hl_hungarian_kernel<<<dim3(B_), 1024, 0, stream>>>(costT, out);
}
